// RSGCNBlock_71107478553042
// MI455X (gfx1250) — compile-verified, matrix operand order checked by reading, not run on hardware
//
#include <hip/hip_runtime.h>
#include <hip/hip_bf16.h>
#include <math.h>

// ---------------------------------------------------------------------------
// RS-GCN on MI455X (gfx1250, wave32). Dense per-edge / per-node matmuls via
// V_WMMA_F32_16X16X32_F16 (f16 A/B in LDS, f32 accumulate).
//  - A/B fragments: two ds_load_b128 per lane (B weights staged transposed)
//  - All gathers/scatters use UNSIGNED 32-bit element offsets so the backend
//    emits saddr+voffset global ops (no per-element 64-bit address math)
//  - global_prefetch of gathered h[src] rows overlapped with the gate WMMAs
//  - Branch-free tanh / sigmoid on v_exp_f32 + v_rcp_f32
// ---------------------------------------------------------------------------

typedef __attribute__((ext_vector_type(16))) _Float16 v16h;
typedef __attribute__((ext_vector_type(8)))  _Float16 v8h;
typedef __attribute__((ext_vector_type(2)))  _Float16 v2h;
typedef __attribute__((ext_vector_type(8)))  float    v8f;

#define HDIM 64
#define XSTRIDE 66   // x row = [pos.x, pos.y, feat[64]]

__device__ __forceinline__ v8f wmma_f16(v16h a, v16h b, v8f c) {
  return __builtin_amdgcn_wmma_f32_16x16x32_f16(
      false, a, false, b, (short)0, c, false, false);
}

__device__ __forceinline__ float fast_exp2(float x) {
#if __has_builtin(__builtin_amdgcn_exp2f)
  return __builtin_amdgcn_exp2f(x);
#else
  return exp2f(x);
#endif
}
__device__ __forceinline__ float fast_rcp(float x) {
#if __has_builtin(__builtin_amdgcn_rcpf)
  return __builtin_amdgcn_rcpf(x);
#else
  return 1.f / x;
#endif
}
__device__ __forceinline__ float fast_tanh(float x) {
  float t = fast_exp2(-2.885390082f * fabsf(x));
  float r = (1.f - t) * fast_rcp(1.f + t);
  return copysignf(r, x);
}
__device__ __forceinline__ float fast_sigmoid(float x) {
  return fast_rcp(1.f + fast_exp2(-1.442695041f * x));
}
__device__ __forceinline__ float dot2acc(v2h a, v2h b, float c) {
#if __has_builtin(__builtin_amdgcn_fdot2)
  return __builtin_amdgcn_fdot2(a, b, c, false);
#else
  return c + (float)a[0] * (float)b[0] + (float)a[1] * (float)b[1];
#endif
}

// 16x32 f16 fragment from a row-major LDS tile (rows indexed by lane%16,
// K-window [k0,k0+32)).  Per CDNA5 ISA the per-lane halves are two
// contiguous 16-byte runs: K = g*8+[0..7] and K = 16+g*8+[0..7].
// Works for A (rows = M) and for B with weights staged transposed.
__device__ __forceinline__ v16h load_frag16(const _Float16* base, int row_stride, int k0) {
  const int lane = threadIdx.x & 31;
  const int r = lane & 15, g = lane >> 4;
  const _Float16* p = base + r * row_stride + k0 + g * 8;
  v8h lo = *(const v8h*)p;
  v8h hi = *(const v8h*)(p + 16);
  return __builtin_shufflevector(lo, hi, 0, 1, 2, 3, 4, 5, 6, 7,
                                 8, 9, 10, 11, 12, 13, 14, 15);
}

// ---------------------------------------------------------------------------
// Fused gate-MLP + message + atomic aggregation (scatter half of one layer).
// Block = 256 threads = 8 waves; each wave owns 16 edges.
// ---------------------------------------------------------------------------
__global__ __launch_bounds__(256) void gate_msg_kernel(
    const float* __restrict__ x,            // (N,66): pos in cols 0..1
    const float* __restrict__ h, int hstride, int hoff,
    const int*   __restrict__ ei,           // (2,E)
    const int*   __restrict__ region,       // (N,)
    const float* __restrict__ Wp1,          // (2,64)
    const float* __restrict__ bp1,          // (64,)
    const float* __restrict__ Wp2,          // (64,64)
    const float* __restrict__ bp2,          // (64,)
    float* __restrict__ agg,                // (N,64) f32, pre-zeroed
    float* __restrict__ deg,                // (N,)   f32, pre-zeroed
    int E)
{
  __shared__ __align__(16) float    sWp1[2 * HDIM];     // f32 (tiny)
  __shared__ __align__(16) float    sbp1[HDIM];
  __shared__ float                  sbp2[HDIM];
  __shared__ __align__(16) _Float16 sWp2t[HDIM * HDIM]; // transposed: [n][k]
  __shared__ __align__(16) _Float16 sHid[8][16 * HDIM];
  __shared__ unsigned               sHofE[8][16];       // element offset into h
  __shared__ unsigned               sAofE[8][16];       // element offset into agg
  __shared__ unsigned               sDofE[8][16];       // element offset into deg
  __shared__ float                  sSame[8][16];

  const int tid  = threadIdx.x;
  const int lane = tid & 31;
  const int wave = tid >> 5;
  const int m    = lane & 15;
  const int g    = lane >> 4;

  if (tid < 2 * HDIM) sWp1[tid] = Wp1[tid];
  if (tid < HDIM) { sbp1[tid] = bp1[tid]; sbp2[tid] = bp2[tid]; }
#pragma unroll
  for (int i = tid; i < HDIM * HDIM; i += 256) {
    const int k = i >> 6, n = i & 63;
    sWp2t[n * HDIM + k] = (_Float16)Wp2[i];   // coalesced read, transpose write
  }

  const int e = blockIdx.x * 128 + wave * 16 + m;

  // Per-row metadata: unsigned 32-bit ELEMENT offsets (arrays < 2^31 bytes).
  if (g == 0) {
    int s = 0, d = 0; float sm = 0.f;
    if (e < E) {
      s = ei[e]; d = ei[E + e];
      sm = (region[s] == region[d]) ? 1.f : 0.f;
    }
    sHofE[wave][m] = (unsigned)(s * hstride + hoff);
    sAofE[wave][m] = (unsigned)(d * HDIM);
    sDofE[wave][m] = (unsigned)d;
    sSame[wave][m] = sm;
  }

  // hidden = relu(rel @ Wp1 + bp1): K=2 on VALU, vector LDS reads/writes.
  float relx = 0.f, rely = 0.f;
  if (e < E) {
    const unsigned s = (unsigned)ei[e] * XSTRIDE;
    const unsigned d = (unsigned)ei[E + e] * XSTRIDE;
    relx = x[s]     - x[d];
    rely = x[s + 1] - x[d + 1];
  }
  __syncthreads();   // weights + metadata visible

  const float4* w0r = (const float4*)&sWp1[0];
  const float4* w1r = (const float4*)&sWp1[HDIM];
  const float4* bbr = (const float4*)sbp1;
#pragma unroll
  for (int v = 0; v < 4; ++v) {
    const int c4 = (g * 32 + v * 8) >> 2;   // float4 index
    const float4 wa0 = w0r[c4], wa1 = w0r[c4 + 1];
    const float4 wb0 = w1r[c4], wb1 = w1r[c4 + 1];
    const float4 ba0 = bbr[c4], ba1 = bbr[c4 + 1];
    v8h pk;
    pk[0] = (_Float16)fmaxf(fmaf(relx, wa0.x, fmaf(rely, wb0.x, ba0.x)), 0.f);
    pk[1] = (_Float16)fmaxf(fmaf(relx, wa0.y, fmaf(rely, wb0.y, ba0.y)), 0.f);
    pk[2] = (_Float16)fmaxf(fmaf(relx, wa0.z, fmaf(rely, wb0.z, ba0.z)), 0.f);
    pk[3] = (_Float16)fmaxf(fmaf(relx, wa0.w, fmaf(rely, wb0.w, ba0.w)), 0.f);
    pk[4] = (_Float16)fmaxf(fmaf(relx, wa1.x, fmaf(rely, wb1.x, ba1.x)), 0.f);
    pk[5] = (_Float16)fmaxf(fmaf(relx, wa1.y, fmaf(rely, wb1.y, ba1.y)), 0.f);
    pk[6] = (_Float16)fmaxf(fmaf(relx, wa1.z, fmaf(rely, wb1.z, ba1.z)), 0.f);
    pk[7] = (_Float16)fmaxf(fmaf(relx, wa1.w, fmaf(rely, wb1.w, ba1.w)), 0.f);
    *(v8h*)&sHid[wave][m * HDIM + g * 32 + v * 8] = pk;
  }
  __syncthreads();

  // Register-cache metadata for the 8 rows this lane scatters (r = g*8+j).
  float smv[8]; unsigned hof[8], aof[8];
#pragma unroll
  for (int j = 0; j < 8; ++j) {
    const int r = g * 8 + j;
    smv[j] = sSame[wave][r];
    hof[j] = sHofE[wave][r];
    aof[j] = sAofE[wave][r];
  }

  // Prefetch the h[src] rows this half-wave will gather (256 B each = 2
  // cachelines; lanes n=0,1 of each half cover offsets 0 and +32 elements),
  // overlapping the gather latency with the gate WMMAs below.
  if ((lane & 15) < 2) {
#pragma unroll
    for (int j = 0; j < 8; ++j) {
      if (smv[j] != 0.f)
        __builtin_prefetch(&h[hof[j] + (unsigned)((lane & 1) * 32)], 0, 1);
    }
  }

  // gatepre[16x64] = hidden @ Wp2 + bp2 via WMMA, fused scatter.
  const v16h a0 = load_frag16(&sHid[wave][0], HDIM, 0);
  const v16h a1 = load_frag16(&sHid[wave][0], HDIM, 32);
  const int  n  = lane & 15;

#pragma unroll
  for (int nt = 0; nt < 4; ++nt) {
    const int n0 = nt * 16;
    v8f c;
    const float bv = sbp2[n0 + n];
#pragma unroll
    for (int j = 0; j < 8; ++j) c[j] = bv;
    c = wmma_f16(a0, load_frag16(sWp2t + n0 * HDIM, HDIM, 0),  c);
    c = wmma_f16(a1, load_frag16(sWp2t + n0 * HDIM, HDIM, 32), c);

    const unsigned ce = (unsigned)(n0 + n);   // column element offset
    // C layout: lanes 0..15 hold rows j, lanes 16..31 rows j+8; col n0+n.
    // Row is uniform per half-wave -> the sm guard skips whole halves.
#pragma unroll
    for (int j = 0; j < 8; ++j) {
      if (smv[j] != 0.f) {
        const float gate = fast_tanh(c[j]);
        const float hv = h[hof[j] + ce];
        atomicAdd(&agg[aof[j] + ce], hv * gate);
      }
    }
  }

  if (g == 0) {
    const float sm = sSame[wave][m];
    if (sm != 0.f)
      atomicAdd(&deg[sDofE[wave][m]], sm);
  }
}

// ---------------------------------------------------------------------------
// Node update: z = relu((agg / max(deg,1)) @ W + b) via WMMA.
// ---------------------------------------------------------------------------
__global__ __launch_bounds__(256) void node_update_kernel(
    const float* __restrict__ agg, const float* __restrict__ deg,
    const float* __restrict__ W, const float* __restrict__ b,
    float* __restrict__ zout, int N)
{
  __shared__ __align__(16) _Float16 sWt[HDIM * HDIM];  // transposed: [n][k]
  __shared__ float                  sb[HDIM];
  __shared__ __align__(16) _Float16 sA[8][16 * HDIM];

  const int tid  = threadIdx.x;
  const int lane = tid & 31;
  const int wave = tid >> 5;
  const int m    = lane & 15;
  const int g    = lane >> 4;

  if (tid < HDIM) sb[tid] = b[tid];
#pragma unroll
  for (int i = tid; i < HDIM * HDIM; i += 256) {
    const int k = i >> 6, n = i & 63;
    sWt[n * HDIM + k] = (_Float16)W[i];
  }

  const int node = blockIdx.x * 128 + wave * 16 + m;
  float invd = 0.f;
  if (node < N) invd = fast_rcp(fmaxf(deg[(unsigned)node], 1.f));
  const float4* arow = (const float4*)agg + (unsigned)node * (HDIM / 4);
#pragma unroll
  for (int q = 0; q < 4; ++q) {
    float4 f0 = {0.f, 0.f, 0.f, 0.f}, f1 = f0;
    if (node < N) { f0 = arow[g * 8 + 2 * q]; f1 = arow[g * 8 + 2 * q + 1]; }
    v8h pk;
    pk[0] = (_Float16)(f0.x * invd); pk[1] = (_Float16)(f0.y * invd);
    pk[2] = (_Float16)(f0.z * invd); pk[3] = (_Float16)(f0.w * invd);
    pk[4] = (_Float16)(f1.x * invd); pk[5] = (_Float16)(f1.y * invd);
    pk[6] = (_Float16)(f1.z * invd); pk[7] = (_Float16)(f1.w * invd);
    *(v8h*)&sA[wave][m * HDIM + g * 32 + q * 8] = pk;
  }
  __syncthreads();

  const v16h a0 = load_frag16(&sA[wave][0], HDIM, 0);
  const v16h a1 = load_frag16(&sA[wave][0], HDIM, 32);
  const int  n  = lane & 15;

#pragma unroll
  for (int nt = 0; nt < 4; ++nt) {
    const int n0 = nt * 16;
    v8f c;
    const float bv = sb[n0 + n];
#pragma unroll
    for (int j = 0; j < 8; ++j) c[j] = bv;
    c = wmma_f16(a0, load_frag16(sWt + n0 * HDIM, HDIM, 0),  c);
    c = wmma_f16(a1, load_frag16(sWt + n0 * HDIM, HDIM, 32), c);
#pragma unroll
    for (int j = 0; j < 8; ++j) {
      const int r  = g * 8 + j;
      const int nd = blockIdx.x * 128 + wave * 16 + r;
      if (nd < N) zout[(unsigned)nd * HDIM + (unsigned)(n0 + n)] = fmaxf(c[j], 0.f);
    }
  }
}

// ---------------------------------------------------------------------------
// Edge decoder: ze=[z[src]||z[dst]] (16x128) -> WMMA K=128 -> relu
//               -> WMMA K=64 -> relu -> dot with Wd3 -> sigmoid.
// ---------------------------------------------------------------------------
__global__ __launch_bounds__(256) void decoder_kernel(
    const float* __restrict__ z, const int* __restrict__ ei,
    const float* __restrict__ Wd1, const float* __restrict__ bd1,
    const float* __restrict__ Wd2, const float* __restrict__ bd2,
    const float* __restrict__ Wd3, const float* __restrict__ bd3,
    float* __restrict__ out, int E)
{
  __shared__ __align__(16) _Float16 sW1t[HDIM * 128];   // transposed, 16 KB
  __shared__ __align__(16) _Float16 sW2t[HDIM * HDIM];  // transposed,  8 KB
  __shared__ __align__(16) _Float16 sW3h[HDIM];
  __shared__ float                  sb1[HDIM];
  __shared__ float                  sb2[HDIM];
  __shared__ __align__(16) _Float16 sZe[8][16 * 128];   // 32 KB; reused hdec1/2

  const int tid  = threadIdx.x;
  const int lane = tid & 31;
  const int wave = tid >> 5;
  const int m    = lane & 15;
  const int g    = lane >> 4;

  if (tid < HDIM) {
    sW3h[tid] = (_Float16)Wd3[tid];
    sb1[tid] = bd1[tid];
    sb2[tid] = bd2[tid];
  }
#pragma unroll
  for (int i = tid; i < 128 * HDIM; i += 256) {
    const int k = i >> 6, n = i & 63;          // Wd1 is (128,64)
    sW1t[n * 128 + k] = (_Float16)Wd1[i];
  }
#pragma unroll
  for (int i = tid; i < HDIM * HDIM; i += 256) {
    const int k = i >> 6, n = i & 63;
    sW2t[n * HDIM + k] = (_Float16)Wd2[i];
  }

  const int e = blockIdx.x * 128 + wave * 16 + m;
  int s = 0, d = 0;
  if (e < E) { s = ei[e]; d = ei[E + e]; }
  const int nidx = (g == 0) ? s : d;  // g=0 stages z[src] (cols 0..63), g=1 z[dst]
  const float4* zrow = (const float4*)z + (unsigned)nidx * (HDIM / 4);
#pragma unroll
  for (int q = 0; q < 8; ++q) {
    const float4 f0 = zrow[2 * q];
    const float4 f1 = zrow[2 * q + 1];
    v8h pk;
    pk[0] = (_Float16)f0.x; pk[1] = (_Float16)f0.y;
    pk[2] = (_Float16)f0.z; pk[3] = (_Float16)f0.w;
    pk[4] = (_Float16)f1.x; pk[5] = (_Float16)f1.y;
    pk[6] = (_Float16)f1.z; pk[7] = (_Float16)f1.w;
    *(v8h*)&sZe[wave][m * 128 + g * 64 + q * 8] = pk;
  }
  __syncthreads();

  // Stage 1: hdec1 = relu(ze @ Wd1 + bd1), K = 128.
  const v16h a0 = load_frag16(&sZe[wave][0], 128, 0);
  const v16h a1 = load_frag16(&sZe[wave][0], 128, 32);
  const v16h a2 = load_frag16(&sZe[wave][0], 128, 64);
  const v16h a3 = load_frag16(&sZe[wave][0], 128, 96);
  const int  n  = lane & 15;
  _Float16* hdec1 = &sZe[wave][0];         // 16x64 (ze already in registers)
  _Float16* hdec2 = &sZe[wave][16 * 64];   // 16x64

#pragma unroll
  for (int nt = 0; nt < 4; ++nt) {
    const int n0 = nt * 16;
    v8f c;
    const float bv = sb1[n0 + n];
#pragma unroll
    for (int j = 0; j < 8; ++j) c[j] = bv;
    c = wmma_f16(a0, load_frag16(sW1t + n0 * 128, 128, 0),  c);
    c = wmma_f16(a1, load_frag16(sW1t + n0 * 128, 128, 32), c);
    c = wmma_f16(a2, load_frag16(sW1t + n0 * 128, 128, 64), c);
    c = wmma_f16(a3, load_frag16(sW1t + n0 * 128, 128, 96), c);
#pragma unroll
    for (int j = 0; j < 8; ++j)
      hdec1[(g * 8 + j) * HDIM + n0 + n] = (_Float16)fmaxf(c[j], 0.f);
  }
  __syncthreads();

  // Stage 2: hdec2 = relu(hdec1 @ Wd2 + bd2), K = 64.
  const v16h h0 = load_frag16(hdec1, HDIM, 0);
  const v16h h1 = load_frag16(hdec1, HDIM, 32);
#pragma unroll
  for (int nt = 0; nt < 4; ++nt) {
    const int n0 = nt * 16;
    v8f c;
    const float bv = sb2[n0 + n];
#pragma unroll
    for (int j = 0; j < 8; ++j) c[j] = bv;
    c = wmma_f16(h0, load_frag16(sW2t + n0 * HDIM, HDIM, 0),  c);
    c = wmma_f16(h1, load_frag16(sW2t + n0 * HDIM, HDIM, 32), c);
#pragma unroll
    for (int j = 0; j < 8; ++j)
      hdec2[(g * 8 + j) * HDIM + n0 + n] = (_Float16)fmaxf(c[j], 0.f);
  }
  __syncthreads();

  // Final: out = sigmoid(hdec2 . Wd3 + bd3), f16 dot2 accumulate per row.
  if (g == 0 && e < E) {
    float sum = bd3[0];
    const v8h* hr = (const v8h*)&hdec2[m * HDIM];
    const v8h* wr = (const v8h*)sW3h;
#pragma unroll
    for (int q = 0; q < 8; ++q) {
      const v8h hv = hr[q];
      const v8h wv = wr[q];
#pragma unroll
      for (int p = 0; p < 4; ++p) {
        v2h ha = {hv[2 * p], hv[2 * p + 1]};
        v2h wa = {wv[2 * p], wv[2 * p + 1]};
        sum = dot2acc(ha, wa, sum);
      }
    }
    out[(unsigned)e] = fast_sigmoid(sum);
  }
}

// ---------------------------------------------------------------------------
// Host orchestration (graph-capture safe: only async ops on `stream`).
// ---------------------------------------------------------------------------
extern "C" void kernel_launch(void* const* d_in, const int* in_sizes, int n_in,
                              void* d_out, int out_size, void* d_ws, size_t ws_size,
                              hipStream_t stream) {
  const float* x      = (const float*)d_in[0];
  const int*   ei     = (const int*)  d_in[1];
  const int*   region = (const int*)  d_in[2];
  const float* Wp1_1  = (const float*)d_in[3];
  const float* bp1_1  = (const float*)d_in[4];
  const float* Wp2_1  = (const float*)d_in[5];
  const float* bp2_1  = (const float*)d_in[6];
  const float* W1     = (const float*)d_in[7];
  const float* b1     = (const float*)d_in[8];
  const float* Wp1_2  = (const float*)d_in[9];
  const float* bp1_2  = (const float*)d_in[10];
  const float* Wp2_2  = (const float*)d_in[11];
  const float* bp2_2  = (const float*)d_in[12];
  const float* W2     = (const float*)d_in[13];
  const float* b2     = (const float*)d_in[14];
  const float* Wd1    = (const float*)d_in[15];
  const float* bd1    = (const float*)d_in[16];
  const float* Wd2    = (const float*)d_in[17];
  const float* bd2    = (const float*)d_in[18];
  const float* Wd3    = (const float*)d_in[19];
  const float* bd3    = (const float*)d_in[20];

  const int N = in_sizes[0] / XSTRIDE;   // x is (N, 66)
  const int E = in_sizes[1] / 2;         // edge_index is (2, E)

  // Workspace layout (f32): [agg N*64][deg N][z1 N*64][z2 N*64]
  float* agg = (float*)d_ws;
  float* deg = agg + (size_t)N * HDIM;
  float* z1  = deg + N;
  float* z2  = z1  + (size_t)N * HDIM;

  const int edgeBlocks = (E + 127) / 128;
  const int nodeBlocks = (N + 127) / 128;
  const size_t aggBytes = ((size_t)N * HDIM + N) * sizeof(float);

  // ---- Layer 1 ----
  hipMemsetAsync(agg, 0, aggBytes, stream);
  gate_msg_kernel<<<edgeBlocks, 256, 0, stream>>>(
      x, /*h=*/x, /*hstride=*/XSTRIDE, /*hoff=*/2, ei, region,
      Wp1_1, bp1_1, Wp2_1, bp2_1, agg, deg, E);
  node_update_kernel<<<nodeBlocks, 256, 0, stream>>>(agg, deg, W1, b1, z1, N);

  // ---- Layer 2 ----
  hipMemsetAsync(agg, 0, aggBytes, stream);
  gate_msg_kernel<<<edgeBlocks, 256, 0, stream>>>(
      x, /*h=*/z1, /*hstride=*/HDIM, /*hoff=*/0, ei, region,
      Wp1_2, bp1_2, Wp2_2, bp2_2, agg, deg, E);
  node_update_kernel<<<nodeBlocks, 256, 0, stream>>>(agg, deg, W2, b2, z2, N);

  // ---- Decoder ----
  decoder_kernel<<<edgeBlocks, 256, 0, stream>>>(
      z2, ei, Wd1, bd1, Wd2, bd2, Wd3, bd3, (float*)d_out, E);
}
